// DynamicHierarchicalAttention_71141838291416
// MI455X (gfx1250) — matrix operand order repaired and checked by reading, not compiled, not run on hardware
//
#include <hip/hip_runtime.h>
#include <hip/hip_bf16.h>
#include <math.h>
#include <stdint.h>

typedef __attribute__((ext_vector_type(16))) _Float16 v16h;
typedef __attribute__((ext_vector_type(8)))  _Float16 v8h;
typedef __attribute__((ext_vector_type(8)))  float    v8f;

namespace dha {

constexpr int Bv  = 2;
constexpr int Tv  = 2048;
constexpr int Dv  = 512;
constexpr int Hv  = 8;
constexpr int HDv = 64;
constexpr int D3v = 3 * Dv;
constexpr int MAXROWS = Bv * Tv;   // 4096

// header slots: 0=n_char(T) 1=m1 2=m2 3=ntok 4=off_charw 5=off_blockw 6=off_globw

// int-region element offsets within workspace
constexpr int HDR_OFF    = 0;
constexpr int IDS1_OFF   = 64;
constexpr int IDS2_OFF   = IDS1_OFF + Tv;
constexpr int TOKMAP_OFF = IDS2_OFF + Tv;
constexpr int S1LO_OFF   = TOKMAP_OFF + Tv;
constexpr int S1HI_OFF   = S1LO_OFF + Tv;
constexpr int S2LO_OFF   = S1HI_OFF + Tv;
constexpr int S2HI_OFF   = S2LO_OFF + Tv;
constexpr int INT_TOTAL  = S2HI_OFF + Tv;

constexpr size_t FBYTES_BASE = (((size_t)INT_TOTAL * 4) + 255) & ~(size_t)255;

// float-region element offsets (from FBYTES_BASE)
constexpr size_t TMPF_OFF   = 0;
constexpr size_t CTX_OFF    = TMPF_OFF + Tv;
constexpr size_t CHARO_OFF  = CTX_OFF + (size_t)MAXROWS * Dv;
constexpr size_t BLOCKS_OFF = CHARO_OFF + (size_t)MAXROWS * Dv;
constexpr size_t BLOCKO_OFF = BLOCKS_OFF + (size_t)MAXROWS * Dv;
constexpr size_t GLOB_OFF   = BLOCKO_OFF + (size_t)MAXROWS * Dv;
constexpr size_t GLOBO_OFF  = GLOB_OFF + (size_t)MAXROWS * Dv;
constexpr size_t FLOAT_TOTAL= GLOBO_OFF + (size_t)MAXROWS * Dv;
// half-region (qkv in f16), after float region
constexpr size_t QKV16_BYTE = FBYTES_BASE + FLOAT_TOTAL * 4;

__device__ __forceinline__ int imind(int a, int b) { return a < b ? a : b; }

__device__ __forceinline__ v8f wmma16(v16h a, v16h b, v8f c) {
  // D = A(16x32 f16) * B(32x16 f16) + C(16x16 f32)
  return __builtin_amdgcn_wmma_f32_16x16x32_f16(false, a, false, b, (short)0, c,
                                                false, false);
}

} // namespace dha

// ---------------------------------------------------------------------------
// Setup: block ids, segment ranges, token gather map, output offsets (serial)
// ---------------------------------------------------------------------------
__global__ void dha_setup_kernel(const float* __restrict__ merge_w,
                                 const float* __restrict__ merge_b,
                                 const float* __restrict__ gmerge_w,
                                 const float* __restrict__ gmerge_b,
                                 int* __restrict__ iw, float* __restrict__ tmpf) {
  using namespace dha;
  if (threadIdx.x != 0 || blockIdx.x != 0) return;
  int* hdr    = iw + HDR_OFF;
  int* ids1   = iw + IDS1_OFF;
  int* ids2   = iw + IDS2_OFF;
  int* tokmap = iw + TOKMAP_OFF;
  int* s1lo = iw + S1LO_OFF; int* s1hi = iw + S1HI_OFF;
  int* s2lo = iw + S2LO_OFF; int* s2hi = iw + S2HI_OFF;

  float c1 = merge_b[0];
  for (int i = 0; i < Dv; ++i) c1 += merge_w[i];
  float run = 0.f;
  for (int t = 0; t < Tv; ++t) { run += c1; tmpf[t] = fmodf(run, 1.0f); }
  int acc = 0;
  for (int t = 0; t < Tv; ++t) {
    float prev = tmpf[(t + Tv - 1) % Tv];
    acc += (tmpf[t] < prev) ? 1 : 0;
    ids1[t] = acc;
  }
  int m1 = ids1[Tv - 1];
  for (int j = 0; j < Tv; ++j) { s1lo[j] = 0; s1hi[j] = 0; }
  for (int t = 0; t < Tv; ++t) {
    int j = ids1[t];
    if (j >= 1) { if (s1hi[j - 1] == 0) s1lo[j - 1] = t; s1hi[j - 1] = t + 1; }
  }

  float c2 = gmerge_b[0];
  for (int i = 0; i < Dv; ++i) c2 += gmerge_w[i];
  run = 0.f;
  for (int t = 0; t < m1; ++t) { run += c2; tmpf[t] = fmodf(run, 1.0f); }
  acc = 0;
  for (int t = 0; t < m1; ++t) {
    float prev = tmpf[(t + m1 - 1) % m1];
    acc += (tmpf[t] < prev) ? 1 : 0;
    ids2[t] = acc;
  }
  int m2 = (m1 > 0) ? ids2[m1 - 1] : 0;
  for (int j = 0; j < Tv; ++j) { s2lo[j] = 0; s2hi[j] = 0; }
  for (int t = 0; t < m1; ++t) {
    int j = ids2[t];
    if (j >= 1) { if (s2hi[j - 1] == 0) s2lo[j - 1] = t; s2hi[j - 1] = t + 1; }
  }

  int ntok = 0;
  for (int t = 0; t < Tv; ++t) {
    int j = ids1[t];
    int tg = (j >= 1) ? ids2[j - 1] : 0;
    if (tg >= 1) tokmap[ntok++] = tg - 1;
  }

  hdr[0] = Tv; hdr[1] = m1; hdr[2] = m2; hdr[3] = ntok;
  hdr[4] = Bv * ntok * Dv;            // expanded ends, char_w starts
  hdr[5] = hdr[4] + Bv * Tv * Tv;     // block_w starts
  hdr[6] = hdr[5] + Bv * m1 * m1;     // glob_w starts
}

// ---------------------------------------------------------------------------
// GEMM: Y(R x N) = X(R x 512) * W(N x 512)^T + bias, one wave per 16x16 tile.
// Output either f32 (Y) or f16 (Y16) -- QKV results are stored f16 for the
// attention stage (halves HBM traffic and removes cvt work in the hot loop).
// ---------------------------------------------------------------------------
__global__ void __launch_bounds__(128)
dha_gemm_kernel(const float* __restrict__ X, const float* __restrict__ W,
                const float* __restrict__ bias, float* __restrict__ Y,
                _Float16* __restrict__ Y16, const int* __restrict__ hdr,
                int nslot, int Ncols) {
  using namespace dha;
  const int Kd = Dv;
  int lane = threadIdx.x & 31;
  int wave = blockIdx.x * (blockDim.x >> 5) + (threadIdx.x >> 5);
  int tilesN = Ncols >> 4;
  int tm = wave / tilesN;
  int tn = wave - tm * tilesN;
  int R = Bv * hdr[nslot];
  if (tm * 16 >= R) return;            // wave-uniform
  int m = lane & 15, g = lane >> 4;
  int rowA = imind(tm * 16 + m, R - 1);
  int colB = tn * 16 + m;
  const float* xp = X + (size_t)rowA * Kd;
  const float* wp = W + (size_t)colB * Kd;
  v8f acc = {0.f, 0.f, 0.f, 0.f, 0.f, 0.f, 0.f, 0.f};
  for (int k0 = 0; k0 < Kd; k0 += 32) {
    v16h a, bf;
#pragma unroll
    for (int p = 0; p < 8; ++p) {      // A: 16x32, ISA f16 A-layout
      int kb = (p < 4) ? (g * 8 + 2 * p) : (16 + g * 8 + 2 * (p - 4));
      float2 t2 = *(const float2*)(xp + k0 + kb);
      a[2 * p] = (_Float16)t2.x; a[2 * p + 1] = (_Float16)t2.y;
    }
#pragma unroll
    for (int p = 0; p < 8; ++p) {      // B: 32x16, B[k][n] = W[n][k]
      int kbb = (p < 4) ? (g * 8 + 2 * p) : (16 + g * 8 + 2 * (p - 4));
      float2 t2 = *(const float2*)(wp + k0 + kbb);
      bf[2 * p] = (_Float16)t2.x; bf[2 * p + 1] = (_Float16)t2.y;
    }
    acc = wmma16(a, bf, acc);
  }
  float bvv = bias[colB];
  bool full = (tm * 16 + 16) <= R;     // uniform fast path: no per-row guards
  if (Y16) {
    _Float16* yp = Y16 + (size_t)(tm * 16 + 8 * g) * Ncols + colB;
    if (full) {
#pragma unroll
      for (int r = 0; r < 8; ++r) yp[(size_t)r * Ncols] = (_Float16)(acc[r] + bvv);
    } else {
#pragma unroll
      for (int r = 0; r < 8; ++r) {
        int row = tm * 16 + r + 8 * g;
        if (row < R) Y16[(size_t)row * Ncols + colB] = (_Float16)(acc[r] + bvv);
      }
    }
  } else {
    float* yp = Y + (size_t)(tm * 16 + 8 * g) * Ncols + colB;
    if (full) {
#pragma unroll
      for (int r = 0; r < 8; ++r) yp[(size_t)r * Ncols] = acc[r] + bvv;
    } else {
#pragma unroll
      for (int r = 0; r < 8; ++r) {
        int row = tm * 16 + r + 8 * g;
        if (row < R) Y[(size_t)row * Ncols + colB] = acc[r] + bvv;
      }
    }
  }
}

// ---------------------------------------------------------------------------
// Attention helpers: score tile from f16 QKV (pure vector loads, no cvt)
// ---------------------------------------------------------------------------
__device__ __forceinline__ v8f dha_score_tile(const _Float16* __restrict__ qkv,
                                              int qrow, int qo, int ko, int rb,
                                              int kt, int n, int m, int g) {
  using namespace dha;
  const int ld = D3v;
  int key = dha::imind(kt + m, n - 1);
  const _Float16* qp = qkv + (size_t)qrow * ld + qo;
  const _Float16* kp = qkv + (size_t)(rb + key) * ld + ko;
  __builtin_prefetch(kp + (size_t)16 * ld, 0, 1);  // next key tile
  v8f c = {0.f, 0.f, 0.f, 0.f, 0.f, 0.f, 0.f, 0.f};
#pragma unroll
  for (int hk = 0; hk < HDv; hk += 32) {
    v8h a0 = *(const v8h*)(qp + hk + g * 8);
    v8h a1 = *(const v8h*)(qp + hk + 16 + g * 8);
    v8h b0 = *(const v8h*)(kp + hk + g * 8);
    v8h b1 = *(const v8h*)(kp + hk + 16 + g * 8);
    v16h a, bf;
#pragma unroll
    for (int e = 0; e < 8; ++e) {
      a[e] = a0[e]; a[e + 8] = a1[e];
      bf[e] = b0[e]; bf[e + 8] = b1[e];
    }
    c = wmma16(a, bf, c);
  }
  return c;
}

// One wave per 16-query tile; loops heads; writes ctx and mean attn weights.
// V tiles for P@V are staged to LDS with gfx1250 async LDS loads.
__global__ void __launch_bounds__(32)
dha_attn_kernel(const _Float16* __restrict__ qkv, float* __restrict__ ctx,
                float* __restrict__ out, const int* __restrict__ hdr,
                int nslot, int offslot) {
  using namespace dha;
  __shared__ float    Pt[16 * 32];     // prob tile restripe buffer
  __shared__ _Float16 Vs[32 * 64];     // async-staged V tile (32 keys x 64 hd)
  int n = hdr[nslot];
  int q0 = blockIdx.x * 16;
  if (q0 >= n) return;
  int b = blockIdx.y;
  int rb = b * n;
  int lane = threadIdx.x & 31;
  int m = lane & 15, g = lane >> 4;
  float* attnw = out + hdr[offslot];
  const float scale = 0.125f;          // 1/sqrt(64)
  const float hinv = 1.0f / Hv;
  int qrow = rb + imind(q0 + m, n - 1);
  unsigned vsbase = (unsigned)(uintptr_t)(&Vs[0]);  // LDS aperture: addr[31:0]

  for (int h = 0; h < Hv; ++h) {
    int qo = h * HDv, ko = Dv + h * HDv, vo = 2 * Dv + h * HDv;

    // ---- pass 1: fused online max + sum-of-exp (per-row stats uniform in
    //      each 16-lane group; row r+8g stats live in group g)
    float mx[8], sm[8];
#pragma unroll
    for (int r = 0; r < 8; ++r) { mx[r] = -1e30f; sm[r] = 0.f; }
    for (int kt = 0; kt < n; kt += 16) {
      v8f c = dha_score_tile(qkv, qrow, qo, ko, rb, kt, n, m, g);
      bool valid = (kt + m) < n;
#pragma unroll
      for (int r = 0; r < 8; ++r) {
        float v = valid ? c[r] * scale : -1e30f;
        float tmax = v;
#pragma unroll
        for (int s = 1; s < 16; s <<= 1) tmax = fmaxf(tmax, __shfl_xor(tmax, s, 16));
        float nmx = fmaxf(mx[r], tmax);
        float e = valid ? __expf(v - nmx) : 0.f;
#pragma unroll
        for (int s = 1; s < 16; s <<= 1) e += __shfl_xor(e, s, 16);
        sm[r] = sm[r] * __expf(mx[r] - nmx) + e;
        mx[r] = nmx;
      }
    }
    float inv[8];
#pragma unroll
    for (int r = 0; r < 8; ++r) inv[r] = 1.0f / sm[r];

    // ---- pass 2: emit probs (mean over heads) and P@V via WMMA
    v8f o[4];
#pragma unroll
    for (int nt = 0; nt < 4; ++nt) o[nt] = (v8f){0.f,0.f,0.f,0.f,0.f,0.f,0.f,0.f};

    for (int kc = 0; kc < n; kc += 32) {
      // stage V tile (32 keys x 64 halfs = 4KB) into LDS asynchronously;
      // wait for prior-iteration LDS reads before overwriting.
      asm volatile("s_wait_dscnt 0x0" ::: "memory");
#pragma unroll
      for (int cc = 0; cc < 8; ++cc) {
        int off = (lane + 32 * cc) * 16;          // byte offset within tile
        int r = off >> 7;                          // key row within tile
        int keyc = imind(kc + r, n - 1);
        const _Float16* src =
            qkv + (size_t)(rb + keyc) * D3v + vo + ((off & 127) >> 1);
        unsigned long long ga = (unsigned long long)(uintptr_t)src;
        unsigned la = vsbase + (unsigned)off;
        asm volatile("global_load_async_to_lds_b128 %0, %1, off"
                     :: "v"(la), "v"(ga) : "memory");
      }

      __syncthreads();
#pragma unroll
      for (int sth = 0; sth < 2; ++sth) {
        int kt = kc + sth * 16;        // uniform
        if (kt < n) {
          v8f c = dha_score_tile(qkv, qrow, qo, ko, rb, kt, n, m, g);
          bool valid = (kt + m) < n;
          int key = kt + m;
#pragma unroll
          for (int r = 0; r < 8; ++r) {
            float p = valid ? __expf(c[r] * scale - mx[r]) * inv[r] : 0.f;
            int qq = q0 + r + 8 * g;
            if (valid && qq < n) {
              float* dst = attnw + (size_t)(rb + qq) * n + key;
              float vput = p * hinv;
              if (h == 0) *dst = vput; else *dst += vput;
            }
            Pt[(r + 8 * g) * 32 + sth * 16 + m] = p;
          }
        } else {
#pragma unroll
          for (int r = 0; r < 8; ++r) Pt[(r + 8 * g) * 32 + sth * 16 + m] = 0.f;
        }
      }
      __syncthreads();
      v16h pa;                         // restripe probs into WMMA A layout
#pragma unroll
      for (int e = 0; e < 16; ++e) {
        int kk = (e < 8) ? (g * 8 + e) : (16 + g * 8 + (e - 8));
        pa[e] = (_Float16)Pt[m * 32 + kk];
      }
      asm volatile("s_wait_asynccnt 0x0" ::: "memory");  // V tile ready in LDS
#pragma unroll
      for (int nt = 0; nt < 4; ++nt) { // HD=64 -> 4 output column tiles
        v16h bv;
#pragma unroll
        for (int e = 0; e < 16; ++e) {
          bv[e] = Vs[((e < 8) ? (g * 8 + e) : (16 + g * 8 + (e - 8))) * 64 + nt * 16 + m];
        }
        o[nt] = wmma16(pa, bv, o[nt]);
      }
    }
    // write ctx (concat heads)
#pragma unroll
    for (int r = 0; r < 8; ++r) {
      int qq = q0 + r + 8 * g;
      if (qq < n) {
        float* cp = ctx + (size_t)(rb + qq) * Dv + h * HDv;
#pragma unroll
        for (int nt = 0; nt < 4; ++nt) cp[nt * 16 + m] = o[nt][r];
      }
    }
  }
}

// ---------------------------------------------------------------------------
// Segment-mean merge and final gather-expand (float4 vectorized)
// ---------------------------------------------------------------------------
__global__ void dha_merge_kernel(const float4* __restrict__ xin,
                                 float4* __restrict__ xout,
                                 const int* __restrict__ slo,
                                 const int* __restrict__ shi,
                                 const int* __restrict__ hdr, int mslot,
                                 int nslot) {
  using namespace dha;
  constexpr int D4 = Dv / 4;
  int idx = blockIdx.x * blockDim.x + threadIdx.x;
  int d = idx & (D4 - 1);
  int j = (idx / D4) & (Tv - 1);
  int b = idx / (D4 * Tv);
  if (b >= Bv) return;
  int mc = hdr[mslot], nin = hdr[nslot];
  if (j >= mc) return;
  int lo = slo[j], hi = shi[j];
  float4 s = make_float4(0.f, 0.f, 0.f, 0.f);
  for (int t = lo; t < hi; ++t) {
    float4 v = xin[(size_t)(b * nin + t) * D4 + d];
    s.x += v.x; s.y += v.y; s.z += v.z; s.w += v.w;
  }
  float r = 1.0f / ((float)(hi - lo) + 1e-10f);
  s.x *= r; s.y *= r; s.z *= r; s.w *= r;
  xout[(size_t)(b * mc + j) * D4 + d] = s;
}

__global__ void dha_expand_kernel(const float4* __restrict__ gout,
                                  const int* __restrict__ tokmap,
                                  const int* __restrict__ hdr,
                                  float4* __restrict__ out) {
  using namespace dha;
  constexpr int D4 = Dv / 4;
  int idx = blockIdx.x * blockDim.x + threadIdx.x;
  int d = idx & (D4 - 1);
  int j = (idx / D4) & (Tv - 1);
  int b = idx / (D4 * Tv);
  if (b >= Bv) return;
  int ntok = hdr[3], m2 = hdr[2];
  if (j >= ntok) return;
  out[(size_t)(b * ntok + j) * D4 + d] =
      gout[(size_t)(b * m2 + tokmap[j]) * D4 + d];
}

// ---------------------------------------------------------------------------
extern "C" void kernel_launch(void* const* d_in, const int* in_sizes, int n_in,
                              void* d_out, int out_size, void* d_ws,
                              size_t ws_size, hipStream_t stream) {
  using namespace dha;
  (void)in_sizes; (void)n_in; (void)out_size; (void)ws_size;

  const float* x        = (const float*)d_in[0];
  const float* char_Wi  = (const float*)d_in[1];
  const float* char_bi  = (const float*)d_in[2];
  const float* char_Wo  = (const float*)d_in[3];
  const float* char_bo  = (const float*)d_in[4];
  const float* block_Wi = (const float*)d_in[5];
  const float* block_bi = (const float*)d_in[6];
  const float* block_Wo = (const float*)d_in[7];
  const float* block_bo = (const float*)d_in[8];
  const float* glob_Wi  = (const float*)d_in[9];
  const float* glob_bi  = (const float*)d_in[10];
  const float* glob_Wo  = (const float*)d_in[11];
  const float* glob_bo  = (const float*)d_in[12];
  const float* merge_w  = (const float*)d_in[13];
  const float* merge_b  = (const float*)d_in[14];
  const float* gmerge_w = (const float*)d_in[15];
  const float* gmerge_b = (const float*)d_in[16];

  int*   iw = (int*)d_ws;
  float* fw = (float*)((char*)d_ws + FBYTES_BASE);
  int* hdr    = iw + HDR_OFF;
  int* tokmap = iw + TOKMAP_OFF;
  int* s1lo = iw + S1LO_OFF; int* s1hi = iw + S1HI_OFF;
  int* s2lo = iw + S2LO_OFF; int* s2hi = iw + S2HI_OFF;
  float* tmpf   = fw + TMPF_OFF;
  float* ctx    = fw + CTX_OFF;
  float* charo  = fw + CHARO_OFF;
  float* blocks = fw + BLOCKS_OFF;
  float* blocko = fw + BLOCKO_OFF;
  float* glob   = fw + GLOB_OFF;
  float* globo  = fw + GLOBO_OFF;
  _Float16* qkv16 = (_Float16*)((char*)d_ws + QKV16_BYTE);
  float* out = (float*)d_out;

  constexpr int GEMM_T      = 128;
  constexpr int QKV_BLOCKS  = (MAXROWS / 16) * (D3v / 16) / 4;  // 6144
  constexpr int PROJ_BLOCKS = (MAXROWS / 16) * (Dv / 16) / 4;   // 2048
  constexpr int ELT4_BLOCKS = (MAXROWS * (Dv / 4)) / 256;       // 2048
  dim3 attn_grid(Tv / 16, Bv);

  dha_setup_kernel<<<1, 1, 0, stream>>>(merge_w, merge_b, gmerge_w, gmerge_b,
                                        iw, tmpf);
  // ---- char level ----
  dha_gemm_kernel<<<QKV_BLOCKS, GEMM_T, 0, stream>>>(x, char_Wi, char_bi,
                                                     nullptr, qkv16, hdr, 0, D3v);
  dha_attn_kernel<<<attn_grid, 32, 0, stream>>>(qkv16, ctx, out, hdr, 0, 4);
  dha_gemm_kernel<<<PROJ_BLOCKS, GEMM_T, 0, stream>>>(ctx, char_Wo, char_bo,
                                                      charo, nullptr, hdr, 0, Dv);
  dha_merge_kernel<<<ELT4_BLOCKS, 256, 0, stream>>>((const float4*)charo,
                                                    (float4*)blocks, s1lo, s1hi,
                                                    hdr, 1, 0);
  // ---- block level ----
  dha_gemm_kernel<<<QKV_BLOCKS, GEMM_T, 0, stream>>>(blocks, block_Wi, block_bi,
                                                     nullptr, qkv16, hdr, 1, D3v);
  dha_attn_kernel<<<attn_grid, 32, 0, stream>>>(qkv16, ctx, out, hdr, 1, 5);
  dha_gemm_kernel<<<PROJ_BLOCKS, GEMM_T, 0, stream>>>(ctx, block_Wo, block_bo,
                                                      blocko, nullptr, hdr, 1, Dv);
  dha_merge_kernel<<<ELT4_BLOCKS, 256, 0, stream>>>((const float4*)blocko,
                                                    (float4*)glob, s2lo, s2hi,
                                                    hdr, 2, 1);
  // ---- global level ----
  dha_gemm_kernel<<<QKV_BLOCKS, GEMM_T, 0, stream>>>(glob, glob_Wi, glob_bi,
                                                     nullptr, qkv16, hdr, 2, D3v);
  dha_attn_kernel<<<attn_grid, 32, 0, stream>>>(qkv16, ctx, out, hdr, 2, 6);
  dha_gemm_kernel<<<PROJ_BLOCKS, GEMM_T, 0, stream>>>(ctx, glob_Wo, glob_bo,
                                                      globo, nullptr, hdr, 2, Dv);
  // ---- expand ----
  dha_expand_kernel<<<ELT4_BLOCKS, 256, 0, stream>>>((const float4*)globo,
                                                     tokmap, hdr, (float4*)out);
}
